// Rudy_56358560858316
// MI455X (gfx1250) — matrix operand order repaired and checked by reading, not compiled, not run on hardware
//
#include <hip/hip_runtime.h>

// RUDY routing-utilization map as a WMMA GEMM:
//   util = Ox^T * Oy,  Ox/Oy = per-net 1D bin overlaps scaled by sqrt(coef),
// K = num_nets, M = N = 256 bins. Ox/Oy slices are generated on the fly into
// LDS (f16) and consumed by v_wmma_f32_16x16x32_f16 with f32 accumulation.

typedef _Float16 half16 __attribute__((ext_vector_type(16)));
typedef float    float8 __attribute__((ext_vector_type(8)));

#define NBX 256
#define NBY 256
#define BSX (1000.0f / 256.0f)   // bin size x
#define BSY (1000.0f / 256.0f)   // bin size y
#define ROWH 40                  // halves per LDS row: 32 nets + 8 pad (80B, 16B aligned)
#define KSTEP 32                 // nets per WMMA K step (16x16x32)
#define NETS_PER_WG 512          // K-chunk per workgroup (16 K steps)

// ---------------- kernel 0: zero the 256x256 output ----------------
__global__ void rudy_zero(float* __restrict__ out, int n) {
    int i = blockIdx.x * blockDim.x + threadIdx.x;
    if (i < n) out[i] = 0.0f;
}

// ---------------- kernel 1: per-net bbox + sqrt(coef) ----------------
__global__ void rudy_bbox(const float* __restrict__ pin_pos,
                          const int*   __restrict__ netpin_start,
                          const int*   __restrict__ flat_netpin,
                          const float* __restrict__ net_w,
                          float* __restrict__ xmin, float* __restrict__ xmax,
                          float* __restrict__ ymin, float* __restrict__ ymax,
                          float* __restrict__ sqc,
                          int numNets, int numPins) {
    int n = blockIdx.x * blockDim.x + threadIdx.x;
    if (n >= numNets) return;
    int s = netpin_start[n];
    int e = netpin_start[n + 1];
    float x0 = 3.4e38f, x1 = -3.4e38f, y0 = 3.4e38f, y1 = -3.4e38f;
    for (int p = s; p < e; ++p) {
        int pi = flat_netpin[p];
        float px = pin_pos[pi];
        float py = pin_pos[numPins + pi];
        x0 = fminf(x0, px); x1 = fmaxf(x1, px);
        y0 = fminf(y0, py); y1 = fmaxf(y1, py);
    }
    if (e <= s) { x0 = x1 = y0 = y1 = 0.0f; }
    float area = fmaxf((x1 - x0) * (y1 - y0), 1e-6f);
    xmin[n] = x0; xmax[n] = x1;
    ymin[n] = y0; ymax[n] = y1;
    sqc[n]  = sqrtf(net_w[n] / area);   // folded into BOTH f16 operands
}

// ---------------- kernel 2: streaming outer-product GEMM via WMMA ----------------
__global__ __launch_bounds__(512, 1)
void rudy_gemm(const float* __restrict__ xmin, const float* __restrict__ xmax,
               const float* __restrict__ ymin, const float* __restrict__ ymax,
               const float* __restrict__ sqc,
               float* __restrict__ out, int numNets) {
    __shared__ float pxl[KSTEP], pxh[KSTEP], pyl[KSTEP], pyh[KSTEP], psq[KSTEP];
    __shared__ alignas(16) _Float16 OxL[NBX * ROWH];   // [bin][k] f16, 20KB
    __shared__ alignas(16) _Float16 OyL[NBY * ROWH];   // [bin][k] f16, 20KB

    const int tid  = threadIdx.x;
    const int lane = tid & 31;
    const int wave = tid >> 5;            // 0..15 -> x-tile index (rows 16w..16w+15)

    const int wgStart = blockIdx.x * NETS_PER_WG;
    int wgEnd = wgStart + NETS_PER_WG;
    if (wgEnd > numNets) wgEnd = numNets;

    // A (16x32 f16) lane layout: lanes 0-15 hold K[0..7]+K[16..23], lanes 16-31 K[8..15]+K[24..31]
    const int mLane  = lane & 15;
    const int hiHalf = lane >> 4;
    const int koffA  = hiHalf * 8;        // half offset for A fragment q0
    const int kbaseB = hiHalf * 8;        // half offset for B fragment (same K layout as A)

    // generation task: one (axis, bin) per thread (512 threads = 2 axes * 256 bins)
    const int   genBin  = tid & 255;
    const int   genAxis = tid >> 8;       // 0 = x, 1 = y
    const float bl = (float)genBin * BSX; // XL = YL = 0
    const float bh = bl + BSX;
    _Float16*   genDst = (genAxis ? OyL : OxL) + genBin * ROWH;
    const float* Plo = genAxis ? pyl : pxl;
    const float* Phi = genAxis ? pyh : pxh;

    float8 acc[16] = {};                  // 16 y-tiles of 16x16 f32 accumulators

    for (int k0 = wgStart; k0 < wgEnd; k0 += KSTEP) {
        __syncthreads();                  // previous step's fragment reads done
        if (tid < KSTEP) {
            int n = k0 + tid;
            if (n < wgEnd) {
                pxl[tid] = xmin[n]; pxh[tid] = xmax[n];
                pyl[tid] = ymin[n]; pyh[tid] = ymax[n];
                psq[tid] = sqc[n];
            } else {
                pxl[tid] = 0.0f; pxh[tid] = 0.0f;
                pyl[tid] = 0.0f; pyh[tid] = 0.0f;
                psq[tid] = 0.0f;          // zero contribution for tail
            }
        }
        __syncthreads();

        // ---- generate 32-net slice of overlaps, f16, into LDS ----
        alignas(16) _Float16 hbuf[KSTEP];
        #pragma unroll
        for (int j = 0; j < KSTEP; ++j) {
            float ov = fminf(Phi[j], bh) - fmaxf(Plo[j], bl);
            ov = fmaxf(ov, 0.0f) * psq[j];
            hbuf[j] = (_Float16)ov;
        }
        {
            const uint4* s4 = (const uint4*)hbuf;
            uint4* d4 = (uint4*)genDst;
            d4[0] = s4[0]; d4[1] = s4[1]; d4[2] = s4[2]; d4[3] = s4[3];
        }
        __syncthreads();

        // ---- A fragment for this wave's x-tile ----
        union { half16 v; uint4 q[2]; } ua;
        const _Float16* arow = OxL + (wave * 16 + mLane) * ROWH + koffA;
        ua.q[0] = *(const uint4*)(arow);        // K koffA..koffA+7
        ua.q[1] = *(const uint4*)(arow + 16);   // K 16+koffA..16+koffA+7

        // ---- sweep all 16 y-tiles ----
        #pragma unroll
        for (int tn = 0; tn < 16; ++tn) {
            union { half16 v; uint4 q[2]; } ub;
            const _Float16* brow = OyL + (tn * 16 + mLane) * ROWH + kbaseB;
            ub.q[0] = *(const uint4*)(brow);      // K kbase..kbase+7
            ub.q[1] = *(const uint4*)(brow + 16); // K 16+kbase..16+kbase+7
            acc[tn] = __builtin_amdgcn_wmma_f32_16x16x32_f16(
                false, ua.v, false, ub.v, (short)0, acc[tn], false, false);
        }
    }

    // ---- epilogue: C/D layout (M = r + 8*hiHalf, N = mLane), scale, atomic add ----
    const float scale = 1.0f / (BSX * BSY);
    #pragma unroll
    for (int tn = 0; tn < 16; ++tn) {
        #pragma unroll
        for (int r = 0; r < 8; ++r) {
            int x = wave * 16 + r + hiHalf * 8;
            int y = tn * 16 + mLane;
            atomicAdd(out + x * NBY + y, acc[tn][r] * scale);
        }
    }
}

extern "C" void kernel_launch(void* const* d_in, const int* in_sizes, int n_in,
                              void* d_out, int out_size, void* d_ws, size_t ws_size,
                              hipStream_t stream) {
    const float* pin_pos      = (const float*)d_in[0];
    const int*   netpin_start = (const int*)d_in[1];
    const int*   flat_netpin  = (const int*)d_in[2];
    const float* net_weights  = (const float*)d_in[3];
    const int numPins = in_sizes[2];
    const int numNets = in_sizes[3];

    float* out = (float*)d_out;
    float* ws  = (float*)d_ws;        // needs 5 * numNets * 4 bytes (= 2 MB here)
    float* xmin = ws;
    float* xmax = ws + (size_t)numNets;
    float* ymin = ws + (size_t)2 * numNets;
    float* ymax = ws + (size_t)3 * numNets;
    float* sqc  = ws + (size_t)4 * numNets;

    rudy_zero<<<(out_size + 255) / 256, 256, 0, stream>>>(out, out_size);
    rudy_bbox<<<(numNets + 255) / 256, 256, 0, stream>>>(
        pin_pos, netpin_start, flat_netpin, net_weights,
        xmin, xmax, ymin, ymax, sqc, numNets, numPins);
    const int grid = (numNets + NETS_PER_WG - 1) / NETS_PER_WG;
    rudy_gemm<<<grid, 512, 0, stream>>>(xmin, xmax, ymin, ymax, sqc, out, numNets);
}
